// MHSelfAttention_46428596470516
// MI455X (gfx1250) — compile-verified, matrix operand order checked by reading, not run on hardware
//
#include <hip/hip_runtime.h>
#include <hip/hip_bf16.h>

// ---------------------------------------------------------------------------
// MI455X (gfx1250) fused multi-head self-attention, f16 WMMA pipeline.
//   1) cvt4:      f32 -> f16 conversion of x and W_{q,k,v}
//   2) qkv_proj:  Y = X * W^T via v_wmma_f32_16x16x32_f16, output [B,H,S,D] f16
//   3) attn:      flash-attention; S^T = K*Q^T WMMA, online softmax,
//                 O^T = V^T * P^T WMMA (V^T staged via LDS), f32 output.
// ---------------------------------------------------------------------------

typedef _Float16 half_t;
typedef __attribute__((ext_vector_type(4)))  _Float16 v4h;
typedef __attribute__((ext_vector_type(8)))  _Float16 v8h;
typedef __attribute__((ext_vector_type(16))) _Float16 v16h;
typedef __attribute__((ext_vector_type(8)))  float    v8f;

#define EMBED 1024
#define SEQ   2048
#define BATCH 4
#define HEADS 16
#define HDIM  64
#define LDV   72   // padded LDS row stride (halves); 144B rows, 16B aligned chunks

__device__ __forceinline__ v8f wmma_f16(v16h a, v16h b, v8f c) {
  // D = A(16x32 f16) * B(32x16 f16) + C(16x16 f32)
  return __builtin_amdgcn_wmma_f32_16x16x32_f16(
      /*neg_a=*/false, a, /*neg_b=*/false, b,
      /*c_mod=*/(short)0, c, /*reuse_a=*/false, /*reuse_b=*/false);
}

// Per-lane A/B operand fragment for 16x16x32 f16 WMMA, loaded straight from a
// row-major matrix whose contraction dim (K) is contiguous.
// Lane layout (ISA 7.12.2): halves 0..7 -> K = k0 + 8*lane_hi + i,
//                           halves 8..15 -> K = k0 + 16 + 8*lane_hi + (i-8).
__device__ __forceinline__ v16h load_frag(const half_t* __restrict__ base,
                                          int row, int ld, int k0, int lane_hi) {
  const half_t* p = base + (size_t)row * ld + k0 + (lane_hi ? 8 : 0);
  v8h lo = *(const v8h*)p;          // 16B global_load_b128
  v8h hi = *(const v8h*)(p + 16);   // 16B global_load_b128
  return __builtin_shufflevector(lo, hi, 0,1,2,3,4,5,6,7,8,9,10,11,12,13,14,15);
}

// --------------------------- f32 -> f16 convert ----------------------------
__global__ void cvt4(const float* __restrict__ src, half_t* __restrict__ dst, int n4) {
  int i = blockIdx.x * blockDim.x + threadIdx.x;
  if (i < n4) {
    float4 f = ((const float4*)src)[i];
    v4h h;
    h[0] = (half_t)f.x; h[1] = (half_t)f.y; h[2] = (half_t)f.z; h[3] = (half_t)f.w;
    ((v4h*)dst)[i] = h;
  }
}

// ------------------------------ QKV projection -----------------------------
// Y[m,n] = sum_k X[m,k] * W[n,k]   (Linear: y = x @ W^T)
// Block = 128 threads = 4 waves; wave owns a 32x64 output tile (2 M-tiles x 4
// N-tiles of 16x16). Output scattered into head-major [B,H,S,D] f16.
__global__ __launch_bounds__(128)
void qkv_proj(const half_t* __restrict__ xh, const half_t* __restrict__ wh,
              half_t* __restrict__ outh) {
  const int wave = threadIdx.x >> 5;
  const int lane = threadIdx.x & 31;
  const int lhi  = lane >> 4;      // which 16-lane half
  const int llo  = lane & 15;
  const int m0 = blockIdx.x * 128 + wave * 32;  // rows (tokens b*S+s)
  const int n0 = blockIdx.y * 64;               // cols (embed e)

  v8f acc[2][4] = {};
  for (int k0 = 0; k0 < EMBED; k0 += 32) {
    v16h a0 = load_frag(xh, m0 + llo,      EMBED, k0, lhi);
    v16h a1 = load_frag(xh, m0 + 16 + llo, EMBED, k0, lhi);
#pragma unroll
    for (int nt = 0; nt < 4; ++nt) {
      // B[k,n] = W[n,k]: lane holds column n -> contiguous row of W
      v16h b = load_frag(wh, n0 + nt * 16 + llo, EMBED, k0, lhi);
      acc[0][nt] = wmma_f16(a0, b, acc[0][nt]);
      acc[1][nt] = wmma_f16(a1, b, acc[1][nt]);
    }
  }
  // C/D layout: VGPR r, lane -> row m = r + 8*lhi, col n = llo
#pragma unroll
  for (int mt = 0; mt < 2; ++mt)
#pragma unroll
    for (int nt = 0; nt < 4; ++nt)
#pragma unroll
      for (int r = 0; r < 8; ++r) {
        int m = m0 + mt * 16 + r + lhi * 8;
        int n = n0 + nt * 16 + llo;
        int b = m >> 11, s = m & (SEQ - 1);
        int h = n >> 6,  d = n & (HDIM - 1);
        outh[(((size_t)(b * HEADS + h) * SEQ) + s) * HDIM + d] = (half_t)acc[mt][nt][r];
      }
}

// ------------------------------- Attention ---------------------------------
// Block = 128 threads = 4 waves, one (b,h), 64 queries (16 per wave).
// Loop over key chunks of 64. Scores computed transposed (S^T = K * Q^T) so
// softmax stats are per-lane scalars and P converts in-register into the
// B-operand of the second WMMA (O^T = V^T * P^T). V^T staged via LDS.
__global__ __launch_bounds__(128)
void attn(const half_t* __restrict__ qh, const half_t* __restrict__ kh,
          const half_t* __restrict__ vh, float* __restrict__ out) {
  __shared__ __align__(16) half_t vt[64 * LDV];   // V^T tile: [d][key]

  const int wave = threadIdx.x >> 5;
  const int lane = threadIdx.x & 31;
  const int lhi  = lane >> 4;
  const int llo  = lane & 15;
  const int bh = blockIdx.y;                  // b*HEADS + h
  const int bb = bh >> 4, hh = bh & 15;
  const int q0 = blockIdx.x * 64 + wave * 16; // wave's query tile base

  const half_t* qbase = qh + (size_t)bh * SEQ * HDIM;
  const half_t* kbase = kh + (size_t)bh * SEQ * HDIM;
  const half_t* vbase = vh + (size_t)bh * SEQ * HDIM;

  // Q as B operand (B[d,q] = Q[q,d]): lane holds column q -> contiguous Q row
  v16h qb[2];
  qb[0] = load_frag(qbase, q0 + llo, HDIM, 0,  lhi);
  qb[1] = load_frag(qbase, q0 + llo, HDIM, 32, lhi);

  float M = -INFINITY, L = 0.f;
  v8f o[4] = {};   // O^T accumulators: d-tiles x (16d x 16q)

  for (int c0 = 0; c0 < SEQ; c0 += 64) {
    // ---- stage V^T chunk (64 keys x 64 d -> vt[d][k]) -----------------
    __syncthreads();
    {
      const half_t* vc = vbase + (size_t)c0 * HDIM;
      int tk = threadIdx.x >> 4;         // 0..7
      int dg = (threadIdx.x & 15) * 4;   // 0,4,...,60
#pragma unroll
      for (int p = 0; p < 8; ++p) {
        int k = p * 8 + tk;
        v4h val = *(const v4h*)(vc + k * HDIM + dg);
#pragma unroll
        for (int j = 0; j < 4; ++j) vt[(dg + j) * LDV + k] = val[j];
      }
    }
    __syncthreads();

    // ---- S^T tiles: rows = keys, cols = queries -----------------------
    v8f sc[4];
#pragma unroll
    for (int t = 0; t < 4; ++t) {
      v8f a = {};
      v16h kA0 = load_frag(kbase, c0 + t * 16 + llo, HDIM, 0,  lhi);
      a = wmma_f16(kA0, qb[0], a);
      v16h kA1 = load_frag(kbase, c0 + t * 16 + llo, HDIM, 32, lhi);
      a = wmma_f16(kA1, qb[1], a);
      sc[t] = a;
    }

    // ---- online softmax (per-lane: one query = column llo) ------------
    float mloc = -INFINITY;
#pragma unroll
    for (int t = 0; t < 4; ++t)
#pragma unroll
      for (int r = 0; r < 8; ++r) {
        float v = sc[t][r] * 0.03125f;   // 1/sqrt(EMBED)
        sc[t][r] = v;
        mloc = fmaxf(mloc, v);
      }
    mloc = fmaxf(mloc, __shfl_xor(mloc, 16, 32));  // lane^16 holds other 8 keys
    float Mn = fmaxf(M, mloc);
    float scale = __expf(M - Mn);
    M = Mn;

    float lloc = 0.f;
#pragma unroll
    for (int t = 0; t < 4; ++t)
#pragma unroll
      for (int r = 0; r < 8; ++r) {
        float p = __expf(sc[t][r] - Mn);
        sc[t][r] = p;
        lloc += p;
      }
    lloc += __shfl_xor(lloc, 16, 32);
    L = L * scale + lloc;
#pragma unroll
    for (int dt = 0; dt < 4; ++dt)
#pragma unroll
      for (int r = 0; r < 8; ++r) o[dt][r] *= scale;

    // ---- P^T -> B operand, purely in-register -------------------------
    // B half i (i<8): key = 32g+i+8*lhi  == sc[2g][i]'s key
    //          (i>=8): key = 32g+16+(i-8)+8*lhi == sc[2g+1][i-8]'s key
    v16h pb[2];
#pragma unroll
    for (int g = 0; g < 2; ++g) {
      v16h t;
#pragma unroll
      for (int i = 0; i < 8; ++i) t[i]     = (half_t)sc[2 * g][i];
#pragma unroll
      for (int i = 0; i < 8; ++i) t[i + 8] = (half_t)sc[2 * g + 1][i];
      pb[g] = t;
    }

    // ---- O^T += V^T(16d x 32k) * P^T(32k x 16q) -----------------------
#pragma unroll
    for (int dt = 0; dt < 4; ++dt)
#pragma unroll
      for (int g = 0; g < 2; ++g) {
        const half_t* lp = &vt[(dt * 16 + llo) * LDV + g * 32 + (lhi ? 8 : 0)];
        v8h lo = *(const v8h*)lp;          // ds_load_b128
        v8h hi = *(const v8h*)(lp + 16);   // ds_load_b128
        v16h aV = __builtin_shufflevector(lo, hi,
                    0,1,2,3,4,5,6,7,8,9,10,11,12,13,14,15);
        o[dt] = wmma_f16(aV, pb[g], o[dt]);
      }
  }

  // ---- normalize + store: out[b, s=q, e=h*64+d] f32 ---------------------
  float inv = 1.0f / L;
  int q = q0 + llo;
#pragma unroll
  for (int dt = 0; dt < 4; ++dt)
#pragma unroll
    for (int r = 0; r < 8; ++r) {
      int d = dt * 16 + r + lhi * 8;
      out[(((size_t)bb * SEQ + q) * EMBED) + hh * HDIM + d] = o[dt][r] * inv;
    }
}

// ------------------------------- launcher ----------------------------------
extern "C" void kernel_launch(void* const* d_in, const int* in_sizes, int n_in,
                              void* d_out, int out_size, void* d_ws, size_t ws_size,
                              hipStream_t stream) {
  const float* x  = (const float*)d_in[0];
  const float* wq = (const float*)d_in[1];
  const float* wk = (const float*)d_in[2];
  const float* wv = (const float*)d_in[3];

  char* ws = (char*)d_ws;
  // Workspace layout (70 MiB total)
  half_t* xh  = (half_t*)(ws);                         // 16 MiB: x f16
  half_t* wqh = (half_t*)(ws + (16ull << 20));         //  2 MiB
  half_t* wkh = (half_t*)(ws + (18ull << 20));         //  2 MiB
  half_t* wvh = (half_t*)(ws + (20ull << 20));         //  2 MiB
  half_t* qhp = (half_t*)(ws + (22ull << 20));         // 16 MiB: Q [B,H,S,D]
  half_t* khp = (half_t*)(ws + (38ull << 20));         // 16 MiB: K
  half_t* vhp = (half_t*)(ws + (54ull << 20));         // 16 MiB: V

  const int nx4 = (BATCH * SEQ * EMBED) / 4;   // 2,097,152
  const int nw4 = (EMBED * EMBED) / 4;         //   262,144
  cvt4<<<(nx4 + 255) / 256, 256, 0, stream>>>(x,  xh,  nx4);
  cvt4<<<(nw4 + 255) / 256, 256, 0, stream>>>(wq, wqh, nw4);
  cvt4<<<(nw4 + 255) / 256, 256, 0, stream>>>(wk, wkh, nw4);
  cvt4<<<(nw4 + 255) / 256, 256, 0, stream>>>(wv, wvh, nw4);

  dim3 gp(BATCH * SEQ / 128, EMBED / 64);      // (64, 16)
  qkv_proj<<<gp, 128, 0, stream>>>(xh, wqh, qhp);
  qkv_proj<<<gp, 128, 0, stream>>>(xh, wkh, khp);
  qkv_proj<<<gp, 128, 0, stream>>>(xh, wvh, vhp);

  dim3 ga(SEQ / 64, BATCH * HEADS);            // (32, 64)
  attn<<<ga, 128, 0, stream>>>(qhp, khp, vhp, (float*)d_out);
}
